// GNNAndGRU_31233002177121
// MI455X (gfx1250) — matrix operand order repaired and checked by reading, not compiled, not run on hardware
//
#include <hip/hip_runtime.h>
#include <hip/hip_bf16.h>
#include <math.h>

typedef __attribute__((ext_vector_type(16))) _Float16 v16h;
typedef __attribute__((ext_vector_type(8)))  _Float16 v8h;
typedef __attribute__((ext_vector_type(8)))  float    v8f;

#define Bsz 4
#define Tt  64
#define Nn  256
#define HG  128
#define Hh  64
#define Ee  4096

__device__ __forceinline__ float lrelu(float x) { return x > 0.f ? x : 0.1f * x; }
__device__ __forceinline__ float sigm(float x)  { return 1.f / (1.f + expf(-x)); }

// ---------------- zero workspace regions (stats/deg/outacc) ----------------
__global__ void zero_kernel(float* p, int n) {
    int i = blockIdx.x * blockDim.x + threadIdx.x;
    if (i < n) p[i] = 0.f;
}

// ---------------- GI = x @ w_ih^T + b_ih  (rows = B*T=256, cols = 384) ------
__global__ void gi_kernel(const float* __restrict__ x, const float* __restrict__ w_ih,
                          const float* __restrict__ b_ih, float* __restrict__ gi) {
    __shared__ float xrow[Nn];
    int row = blockIdx.x;                  // b*T + t
    int tid = threadIdx.x;                 // 384 threads
    if (tid < Nn) xrow[tid] = x[row * Nn + tid];
    __syncthreads();
    float acc = b_ih[tid];
    const float* w = w_ih + tid * Nn;
#pragma unroll 8
    for (int k = 0; k < Nn; ++k) acc += xrow[k] * w[k];
    gi[row * (3 * HG) + tid] = acc;
}

// ---------------- sequential GRU over T; writes hs (B,T,HG) ----------------
// w_hh (384x128) staged transposed in LDS: whhT[k][j], bank-conflict-free in j.
__global__ void __launch_bounds__(512) gru_kernel(
    const float* __restrict__ gi, const float* __restrict__ w_hh,
    const float* __restrict__ b_hh, float* __restrict__ hs) {
    __shared__ float whhT[HG * 3 * HG];    // 128 * 384 floats = 192KB
    __shared__ float h[Bsz * HG];
    int tid = threadIdx.x;                 // 512 = 4*128
    for (int idx = tid; idx < HG * 3 * HG; idx += 512) {
        int k = idx / (3 * HG), j = idx % (3 * HG);
        whhT[idx] = w_hh[j * HG + k];
    }
    int b = tid >> 7, hh = tid & 127;
    h[tid] = 0.f;
    __syncthreads();
    float bhr = b_hh[hh], bhz = b_hh[HG + hh], bhn = b_hh[2 * HG + hh];
    for (int t = 0; t < Tt; ++t) {
        const float* hb = h + b * HG;
        float ghr = bhr, ghz = bhz, ghn = bhn;
#pragma unroll 4
        for (int k = 0; k < HG; ++k) {
            float hv = hb[k];
            const float* wrow = whhT + k * (3 * HG);
            ghr += hv * wrow[hh];
            ghz += hv * wrow[HG + hh];
            ghn += hv * wrow[2 * HG + hh];
        }
        const float* gir = gi + (b * Tt + t) * (3 * HG);
        float r = sigm(gir[hh] + ghr);
        float z = sigm(gir[HG + hh] + ghz);
        float n = tanhf(gir[2 * HG + hh] + r * ghn);
        float hnew = (1.f - z) * n + z * hb[hh];
        __syncthreads();
        h[tid] = hnew;
        __syncthreads();
        hs[(b * Tt + t) * HG + hh] = hnew;
    }
}

// ------------- xs = hs @ lin_w^T + lin_b; store transposed xt (B,N,T) ------
__global__ void lin_kernel(const float* __restrict__ hs, const float* __restrict__ lin_w,
                           const float* __restrict__ lin_b, float* __restrict__ xt) {
    __shared__ float hrow[HG];
    int row = blockIdx.x;                  // b*T + t
    int b = row >> 6, t = row & 63;
    int n = threadIdx.x;                   // 256
    if (n < HG) hrow[n] = hs[row * HG + n];
    __syncthreads();
    float acc = lin_b[n];
    const float* w = lin_w + n * HG;
#pragma unroll 4
    for (int k = 0; k < HG; ++k) acc += hrow[k] * w[k];
    xt[(b * Nn + n) * Tt + t] = acc;
}

// ---------------- fused pairwise MLP: GEMM1(d,p->128)+lrelu, GEMM2(->64) ---
// 4 waves/block, each wave loops over 4 tiles of 16 pairs (fixed i, 16 j's).
// Weights staged once per block -> 16x better L2 amortization than 1 tile.
__global__ void __launch_bounds__(128) pair_kernel(
    const float* __restrict__ xt,
    const float* __restrict__ f1w, const float* __restrict__ f1b,
    const float* __restrict__ f2w, const float* __restrict__ f2b,
    const float* __restrict__ a1w, const float* __restrict__ a1b,
    float* __restrict__ y1, float* __restrict__ stats1) {
    __shared__ alignas(32) _Float16 wT1[64 * 128];     // f1_w^T [k][n]
    __shared__ alignas(32) _Float16 wT2[64 * 128];     // f2_w^T
    __shared__ alignas(32) _Float16 aT1[128 * 64];     // a1_w^T
    __shared__ alignas(32) _Float16 stage[4][16 * 128];
    __shared__ float statbuf[128];
    int tid = threadIdx.x;
    for (int idx = tid; idx < 64 * 128; idx += 128) {
        int k = idx >> 7, n = idx & 127;
        wT1[idx] = (_Float16)f1w[n * 64 + k];
        wT2[idx] = (_Float16)f2w[n * 64 + k];
    }
    for (int idx = tid; idx < 128 * 64; idx += 128) {
        int k = idx >> 6, n = idx & 63;
        aT1[idx] = (_Float16)a1w[n * 128 + k];
    }
    if (tid < 128) statbuf[tid] = 0.f;
    __syncthreads();

    int wave = tid >> 5, lane = tid & 31;
    int g = lane >> 4, nl = lane & 15, m = lane & 15;
    v8f zf = {0.f, 0.f, 0.f, 0.f, 0.f, 0.f, 0.f, 0.f};

    for (int it = 0; it < 4; ++it) {
        int gt = blockIdx.x * 16 + wave * 4 + it;  // 16384 tiles = 4 * 256 * 16
        int b = gt >> 12, rem = gt & 4095;
        int i = rem >> 4, j0 = (rem & 15) << 4;

        const float* xi = xt + (b * Nn + i) * Tt;
        const float* xj = xt + (b * Nn + j0 + m) * Tt;

        // Build A operands (d and p), K = 64 => two 32-wide K blocks
        v16h dA[2], pA[2];
#pragma unroll
        for (int kb = 0; kb < 2; ++kb)
#pragma unroll
            for (int hv = 0; hv < 16; ++hv) {
                int K = kb * 32 + (hv & 7) + ((hv >> 3) << 4) + (g << 3);
                float a = xi[K], c = xj[K];
                dA[kb][hv] = (_Float16)fabsf(a - c);
                pA[kb][hv] = (_Float16)(a * c);
            }

        v8f c1[8], c2[8];
#pragma unroll
        for (int nt = 0; nt < 8; ++nt) { c1[nt] = zf; c2[nt] = zf; }
#pragma unroll
        for (int kb = 0; kb < 2; ++kb) {
#pragma unroll
            for (int nt = 0; nt < 8; ++nt) {
                v16h b1, b2;
#pragma unroll
                for (int s = 0; s < 16; ++s) {
                    int K = kb * 32 + (s & 7) + ((s >> 3) << 4) + (g << 3);
                    b1[s] = wT1[K * 128 + nt * 16 + nl];
                    b2[s] = wT2[K * 128 + nt * 16 + nl];
                }
                c1[nt] = __builtin_amdgcn_wmma_f32_16x16x32_f16(false, dA[kb], false, b1,
                                                                (short)0, c1[nt], false, false);
                c2[nt] = __builtin_amdgcn_wmma_f32_16x16x32_f16(false, pA[kb], false, b2,
                                                                (short)0, c2[nt], false, false);
            }
        }
        // epilogue1: bias + leaky-relu + add, stage as f16 for GEMM2 A operand
#pragma unroll
        for (int nt = 0; nt < 8; ++nt) {
            int n = nt * 16 + nl;
            float bb1 = f1b[n], bb2 = f2b[n];
#pragma unroll
            for (int r = 0; r < 8; ++r) {
                int mm = r + 8 * g;
                stage[wave][mm * 128 + n] =
                    (_Float16)(lrelu(c1[nt][r] + bb1) + lrelu(c2[nt][r] + bb2));
            }
        }
        __syncthreads();   // uniform across all waves (same trip count)

        // GEMM2: t1(16x128) @ a1_w^T(128x64)
        v8f y[4];
#pragma unroll
        for (int nt = 0; nt < 4; ++nt) y[nt] = zf;
#pragma unroll
        for (int kb = 0; kb < 4; ++kb) {
            v8h lo = *(const v8h*)&stage[wave][m * 128 + kb * 32 + g * 8];
            v8h hi = *(const v8h*)&stage[wave][m * 128 + kb * 32 + 16 + g * 8];
            v16h A;
#pragma unroll
            for (int q = 0; q < 8; ++q) { A[q] = lo[q]; A[q + 8] = hi[q]; }
#pragma unroll
            for (int nt = 0; nt < 4; ++nt) {
                v16h Bm;
#pragma unroll
                for (int s = 0; s < 16; ++s)
                    Bm[s] = aT1[(kb * 32 + (s & 7) + ((s >> 3) << 4) + (g << 3)) * 64 + nt * 16 + nl];
                y[nt] = __builtin_amdgcn_wmma_f32_16x16x32_f16(false, A, false, Bm,
                                                               (short)0, y[nt], false, false);
            }
        }
        // epilogue2: + a1_b, write y1, accumulate BN1 stats
        long rowbase = (long)(b * Nn + i) * Nn + j0;
#pragma unroll
        for (int nt = 0; nt < 4; ++nt) {
            int n = nt * 16 + nl;
            float bb = a1b[n], s = 0.f, sq = 0.f;
#pragma unroll
            for (int r = 0; r < 8; ++r) {
                int mm = r + 8 * g;
                float v = y[nt][r] + bb;
                y1[(rowbase + mm) * 64 + n] = v;
                s += v; sq += v * v;
            }
            atomicAdd(&statbuf[n], s);
            atomicAdd(&statbuf[64 + n], sq);
        }
        __syncthreads();   // protect stage reuse next iteration (uniform)
    }
    if (tid < 128) atomicAdd(&stats1[tid], statbuf[tid]);
}

// ---------------- BN finalize: scale/shift from sums --------------------
__global__ void bnfin_kernel(const float* __restrict__ stats, const float* __restrict__ g,
                             const float* __restrict__ bta, float* __restrict__ bnp,
                             float inv_count) {
    int n = threadIdx.x;   // 64
    float mean = stats[n] * inv_count;
    float var  = stats[64 + n] * inv_count - mean * mean;
    float sc   = g[n] * rsqrtf(var + 1e-5f);
    bnp[n]      = sc;
    bnp[64 + n] = bta[n] - mean * sc;
}

// ---------------- pass2: bn1+lrelu then @ a2_w^T (in-place) --------------
// 4 waves/block x 4 row-tiles per wave.
__global__ void __launch_bounds__(128) pass2_kernel(
    float* __restrict__ y, const float* __restrict__ a2w, const float* __restrict__ a2b,
    const float* __restrict__ bnp1, float* __restrict__ stats2) {
    __shared__ alignas(32) _Float16 aT[64 * 64];
    __shared__ float sc[64], sh[64];
    __shared__ float statbuf[128];
    int tid = threadIdx.x;
    for (int idx = tid; idx < 64 * 64; idx += 128) {
        int k = idx >> 6, n = idx & 63;
        aT[idx] = (_Float16)a2w[n * 64 + k];
    }
    if (tid < 64) { sc[tid] = bnp1[tid]; sh[tid] = bnp1[64 + tid]; }
    if (tid < 128) statbuf[tid] = 0.f;
    __syncthreads();

    int wave = tid >> 5, lane = tid & 31;
    int g = lane >> 4, nl = lane & 15, m = lane & 15;
    v8f zf = {0.f, 0.f, 0.f, 0.f, 0.f, 0.f, 0.f, 0.f};

    for (int it = 0; it < 4; ++it) {
        long row0 = ((long)blockIdx.x * 16 + wave * 4 + it) * 16;  // 16384 tiles

        v16h A[2];
#pragma unroll
        for (int kb = 0; kb < 2; ++kb)
#pragma unroll
            for (int hv = 0; hv < 16; ++hv) {
                int K = kb * 32 + (hv & 7) + ((hv >> 3) << 4) + (g << 3);
                float v = y[(row0 + m) * 64 + K] * sc[K] + sh[K];
                A[kb][hv] = (_Float16)lrelu(v);
            }

        v8f acc[4];
#pragma unroll
        for (int nt = 0; nt < 4; ++nt) acc[nt] = zf;
#pragma unroll
        for (int kb = 0; kb < 2; ++kb)
#pragma unroll
            for (int nt = 0; nt < 4; ++nt) {
                v16h Bm;
#pragma unroll
                for (int s = 0; s < 16; ++s)
                    Bm[s] = aT[(kb * 32 + (s & 7) + ((s >> 3) << 4) + (g << 3)) * 64 + nt * 16 + nl];
                acc[nt] = __builtin_amdgcn_wmma_f32_16x16x32_f16(false, A[kb], false, Bm,
                                                                 (short)0, acc[nt], false, false);
            }
#pragma unroll
        for (int nt = 0; nt < 4; ++nt) {
            int n = nt * 16 + nl;
            float bb = a2b[n], s = 0.f, sq = 0.f;
#pragma unroll
            for (int r = 0; r < 8; ++r) {
                int mm = r + 8 * g;
                float v = acc[nt][r] + bb;
                y[(row0 + mm) * 64 + n] = v;
                s += v; sq += v * v;
            }
            atomicAdd(&statbuf[n], s);
            atomicAdd(&statbuf[64 + n], sq);
        }
    }
    __syncthreads();
    if (tid < 128) atomicAdd(&stats2[tid], statbuf[tid]);
}

// ---------------- bn2 + lrelu + dot(al_w) + al_b -> a_raw ----------------
__global__ void finala_kernel(const float* __restrict__ y, const float* __restrict__ bnp2,
                              const float* __restrict__ alw, const float* __restrict__ alb,
                              float* __restrict__ araw) {
    long row = (long)blockIdx.x * blockDim.x + threadIdx.x;  // 262144 rows
    float acc = alb[0];
#pragma unroll 4
    for (int n = 0; n < 64; ++n) {
        float v = y[row * 64 + n] * bnp2[n] + bnp2[64 + n];
        acc += lrelu(v) * alw[n];
    }
    araw[row] = acc;
}

// ---------------- a_hat = sigmoid(0.5*(a + a^T)) -> d_out[2nd half] ------
__global__ void ahat_kernel(const float* __restrict__ araw, float* __restrict__ out2) {
    int idx = blockIdx.x * blockDim.x + threadIdx.x;  // 262144
    int b = idx >> 16, r = (idx >> 8) & 255, c = idx & 255;
    float v = 0.5f * (araw[idx] + araw[b * 65536 + c * 256 + r]);
    out2[idx] = sigm(v);
}

// ---------------- xl = xt @ gcn_w^T  (4*256 rows x 256, K=64) ------------
__global__ void xl_kernel(const float* __restrict__ xt, const float* __restrict__ gcnw,
                          float* __restrict__ xl) {
    __shared__ float xrow[Tt];
    int row = blockIdx.x;                 // b*N + i
    int o = threadIdx.x;                  // 256
    if (o < Tt) xrow[o] = xt[row * Tt + o];
    __syncthreads();
    float acc = 0.f;
    const float* w = gcnw + o * Tt;
#pragma unroll 4
    for (int k = 0; k < Tt; ++k) acc += xrow[k] * w[k];
    xl[row * Nn + o] = acc;
}

// ---------------- degree scatter -----------------------------------------
__global__ void deg_kernel(const float* __restrict__ ahat, const int* __restrict__ edge,
                           float* __restrict__ deg) {
    int idx = blockIdx.x * blockDim.x + threadIdx.x;   // 4*4096
    if (idx >= Bsz * Ee) return;
    int b = idx >> 12, e = idx & (Ee - 1);
    int r = edge[e], c = edge[Ee + e];
    float ew = ahat[b * 65536 + r * 256 + c];
    atomicAdd(&deg[b * Nn + c], ew);
}

// ---------------- message scatter ----------------------------------------
__global__ void msg_kernel(const float* __restrict__ ahat, const int* __restrict__ edge,
                           const float* __restrict__ deg, const float* __restrict__ xl,
                           float* __restrict__ outacc) {
    int be = blockIdx.x;                   // 4*4096 blocks
    int b = be >> 12, e = be & (Ee - 1);
    int r = edge[e], c = edge[Ee + e];
    float ew = ahat[b * 65536 + r * 256 + c];
    float dr = deg[b * Nn + r], dc = deg[b * Nn + c];
    float disr = dr > 0.f ? rsqrtf(dr) : 0.f;
    float disc = dc > 0.f ? rsqrtf(dc) : 0.f;
    float norm = disr * ew * disc;
    int o = threadIdx.x;                   // 256
    atomicAdd(&outacc[(b * Nn + c) * Nn + o], norm * xl[(b * Nn + r) * Nn + o]);
}

// ---------------- preds = swap(out,1,2) + gcn_b -> d_out[1st half] -------
__global__ void preds_kernel(const float* __restrict__ outacc, const float* __restrict__ gcnb,
                             float* __restrict__ out) {
    int idx = blockIdx.x * blockDim.x + threadIdx.x;   // 262144
    int b = idx >> 16, o = (idx >> 8) & 255, n = idx & 255;
    out[idx] = outacc[(b * Nn + n) * Nn + o] + gcnb[o];
}

extern "C" void kernel_launch(void* const* d_in, const int* in_sizes, int n_in,
                              void* d_out, int out_size, void* d_ws, size_t ws_size,
                              hipStream_t stream) {
    const float* x      = (const float*)d_in[0];
    const float* gwih   = (const float*)d_in[1];
    const float* gwhh   = (const float*)d_in[2];
    const float* gbih   = (const float*)d_in[3];
    const float* gbhh   = (const float*)d_in[4];
    const float* linw   = (const float*)d_in[5];
    const float* linb   = (const float*)d_in[6];
    const float* f1w    = (const float*)d_in[7];
    const float* f1b    = (const float*)d_in[8];
    const float* f2w    = (const float*)d_in[9];
    const float* f2b    = (const float*)d_in[10];
    const float* a1w    = (const float*)d_in[11];
    const float* a1b    = (const float*)d_in[12];
    const float* bn1g   = (const float*)d_in[13];
    const float* bn1b   = (const float*)d_in[14];
    const float* a2w    = (const float*)d_in[15];
    const float* a2b    = (const float*)d_in[16];
    const float* bn2g   = (const float*)d_in[17];
    const float* bn2b   = (const float*)d_in[18];
    const float* alw    = (const float*)d_in[19];
    const float* alb    = (const float*)d_in[20];
    const float* gcnw   = (const float*)d_in[21];
    const float* gcnb   = (const float*)d_in[22];
    const int*   edge   = (const int*)d_in[23];

    float* ws = (float*)d_ws;
    float* gi     = ws;                       // 98304
    float* hs     = gi + 98304;               // 32768
    float* xt     = hs + 32768;               // 65536
    float* y1     = xt + 65536;               // 16777216
    float* araw   = y1 + 16777216;            // 262144
    float* xl     = araw + 262144;            // 262144
    float* outacc = xl + 262144;              // 262144
    float* deg    = outacc + 262144;          // 1024
    float* stats1 = deg + 1024;               // 128
    float* stats2 = stats1 + 128;             // 128
    float* bnp1   = stats2 + 128;             // 128
    float* bnp2   = bnp1 + 128;               // 128

    float* out = (float*)d_out;
    float* out_preds = out;
    float* out_ahat  = out + 262144;

    // zero: outacc + deg + stats1 + stats2 (contiguous region, 263424 floats)
    zero_kernel<<<(263424 + 255) / 256, 256, 0, stream>>>(outacc, 263424);

    gi_kernel<<<Bsz * Tt, 3 * HG, 0, stream>>>(x, gwih, gbih, gi);
    gru_kernel<<<1, Bsz * HG, 0, stream>>>(gi, gwhh, gbhh, hs);
    lin_kernel<<<Bsz * Tt, Nn, 0, stream>>>(hs, linw, linb, xt);

    pair_kernel<<<1024, 128, 0, stream>>>(xt, f1w, f1b, f2w, f2b, a1w, a1b, y1, stats1);
    bnfin_kernel<<<1, 64, 0, stream>>>(stats1, bn1g, bn1b, bnp1, 1.f / 262144.f);
    pass2_kernel<<<1024, 128, 0, stream>>>(y1, a2w, a2b, bnp1, stats2);
    bnfin_kernel<<<1, 64, 0, stream>>>(stats2, bn2g, bn2b, bnp2, 1.f / 262144.f);
    finala_kernel<<<1024, 256, 0, stream>>>(y1, bnp2, alw, alb, araw);
    ahat_kernel<<<1024, 256, 0, stream>>>(araw, out_ahat);

    xl_kernel<<<Bsz * Nn, Nn, 0, stream>>>(xt, gcnw, xl);
    deg_kernel<<<(Bsz * Ee + 255) / 256, 256, 0, stream>>>(out_ahat, edge, deg);
    msg_kernel<<<Bsz * Ee, Nn, 0, stream>>>(out_ahat, edge, deg, xl, outacc);
    preds_kernel<<<1024, 256, 0, stream>>>(outacc, gcnb, out_preds);
}
